// GAT_59811714564277
// MI455X (gfx1250) — matrix operand order repaired and checked by reading, not compiled, not run on hardware
//
#include <hip/hip_runtime.h>
#include <hip/hip_bf16.h>
#include <math.h>

// ---------------- types for WMMA ----------------
typedef __bf16 bf16x16 __attribute__((ext_vector_type(16)));
typedef float  f32x8   __attribute__((ext_vector_type(8)));

#define IN_CH 500
#define D1    128   // HID*HEADS
#define KT_N  16    // ceil(500/32)

// ---------------- helpers ----------------
__device__ __forceinline__ unsigned f2ord(float f) {
    unsigned u = __float_as_uint(f);
    return (u & 0x80000000u) ? ~u : (u | 0x80000000u);
}
__device__ __forceinline__ float ord2f(unsigned e) {
    unsigned u = (e & 0x80000000u) ? (e ^ 0x80000000u) : ~e;
    return __uint_as_float(u);
}
__device__ __forceinline__ float leaky(float x) { return x >= 0.f ? x : 0.2f * x; }
__device__ __forceinline__ void edge_sd(const int* __restrict__ ei, int E, int e, int& s, int& d) {
    if (e < E) { s = ei[e]; d = ei[E + e]; } else { s = d = e - E; }  // self loops appended
}

// ---------------- zero fill ----------------
__global__ void k_zero_f32(float* p, int n) {
    int i = blockIdx.x * blockDim.x + threadIdx.x;
    if (i < n) p[i] = 0.f;
}

// ---------------- pack W1 into WMMA B-fragment layout (bf16) ----------------
// B frag (32x16, K-major): lane l holds column n = nt*16 + (l&15), K runs of 8
// at kt*32 + (l>>4)*8 and kt*32 + 16 + (l>>4)*8.  Bp[((kt*8+nt)*32 + lane)*16 + j]
__global__ void k_pack_w1(const float* __restrict__ W1, __bf16* __restrict__ Bp) {
    int frag = blockIdx.x;            // kt*8 + nt, 128 frags
    int kt = frag >> 3, nt = frag & 7;
    int lane = threadIdx.x;           // 32 threads
    int n  = nt * 16 + (lane & 15);
    int kb = kt * 32 + (lane >> 4) * 8;
    __bf16* dst = Bp + ((size_t)frag * 32 + lane) * 16;
#pragma unroll
    for (int j = 0; j < 16; ++j) {
        int k = kb + (j < 8 ? j : 8 + j);
        float v = (k < IN_CH) ? W1[(size_t)k * D1 + n] : 0.f;
        dst[j] = (__bf16)v;
    }
}

// ---------------- GEMM1: h1[N,128] = x[N,500] @ W1[500,128] via bf16 WMMA ----
__global__ __launch_bounds__(256) void k_gemm1(const float* __restrict__ x,
                                               const __bf16* __restrict__ Bp,
                                               float* __restrict__ h1, int N) {
    int lane = threadIdx.x & 31;
    int wave = threadIdx.x >> 5;
    int rowBase = blockIdx.x * 128 + wave * 16;
    int half = lane >> 4;       // 0/1: which K half this lane carries
    int m    = lane & 15;
    int row  = rowBase + m;
    int rc   = row < N ? row : N - 1;
    const float* xr = x + (size_t)rc * IN_CH;

    f32x8 acc[8] = {};
    const bf16x16* bpv = reinterpret_cast<const bf16x16*>(Bp);

    for (int kt = 0; kt < KT_N; ++kt) {
        // A fragment (16x32 bf16): elems 0..7 -> K = kt*32 + half*8 + j,
        //                          elems 8..15 -> K = kt*32 + 16 + half*8 + j
        bf16x16 a;
        int kb = kt * 32 + half * 8;
        if (kt < KT_N - 1) {
#pragma unroll
            for (int j = 0; j < 8; ++j) a[j] = (__bf16)xr[kb + j];
#pragma unroll
            for (int j = 0; j < 8; ++j) a[8 + j] = (__bf16)xr[kb + 16 + j];
        } else {
#pragma unroll
            for (int j = 0; j < 8; ++j) { int k = kb + j;      a[j]     = (__bf16)(k < IN_CH ? xr[k] : 0.f); }
#pragma unroll
            for (int j = 0; j < 8; ++j) { int k = kb + 16 + j; a[8 + j] = (__bf16)(k < IN_CH ? xr[k] : 0.f); }
        }
        const bf16x16* bk = bpv + ((size_t)kt * 8) * 32 + lane;
#pragma unroll
        for (int nt = 0; nt < 8; ++nt) {
            bf16x16 b = bk[nt * 32];
            acc[nt] = __builtin_amdgcn_wmma_f32_16x16x32_bf16(
                false, a, false, b, (short)0, acc[nt], false, false);
        }
    }
    // C/D layout: lane holds col nt*16+m; vgpr j -> row rowBase + half*8 + j
#pragma unroll
    for (int nt = 0; nt < 8; ++nt) {
        int col = nt * 16 + m;
#pragma unroll
        for (int j = 0; j < 8; ++j) {
            int r = rowBase + half * 8 + j;
            if (r < N) h1[(size_t)r * D1 + col] = acc[nt][j];
        }
    }
}

// ---------------- per-node attention projections, layer 1 -------------------
__global__ void k_proj1(const float* __restrict__ h1,
                        const float* __restrict__ att_src, const float* __restrict__ att_dst,
                        float* a_src, float* a_dst, unsigned* nmax, float* nsum, int N) {
    int id = blockIdx.x * blockDim.x + threadIdx.x;   // node*4 + head
    if (id >= N * 4) return;
    int node = id >> 2, head = id & 3;
    const float* hp = h1 + (size_t)node * D1 + head * 32;
    const float* as = att_src + head * 32;
    const float* ad = att_dst + head * 32;
    float s = 0.f, d = 0.f;
#pragma unroll 8
    for (int c = 0; c < 32; ++c) { float v = hp[c]; s += v * as[c]; d += v * ad[c]; }
    a_src[id] = s; a_dst[id] = d;
    nmax[id] = 0u;       // < encoding of every real float (enc(-inf)=0x007FFFFF)
    nsum[id] = 0.f;
}

// ---------------- layer-1 edge passes (H=4) ---------------------------------
__global__ void k_edge1_max(const int* __restrict__ ei, int E, int Et,
                            const float* __restrict__ a_s, const float* __restrict__ a_d,
                            unsigned* nmax) {
    int e = blockIdx.x * blockDim.x + threadIdx.x;
    if (e >= Et) return;
    int s, d; edge_sd(ei, E, e, s, d);
#pragma unroll
    for (int h = 0; h < 4; ++h) {
        float l = leaky(a_s[s * 4 + h] + a_d[d * 4 + h]);
        atomicMax(&nmax[d * 4 + h], f2ord(l));
    }
}
__global__ void k_edge1_exp(const int* __restrict__ ei, int E, int Et,
                            const float* __restrict__ a_s, const float* __restrict__ a_d,
                            const unsigned* __restrict__ nmax, float* nsum, float* alpha) {
    int e = blockIdx.x * blockDim.x + threadIdx.x;
    if (e >= Et) return;
    int s, d; edge_sd(ei, E, e, s, d);
#pragma unroll
    for (int h = 0; h < 4; ++h) {
        float l  = leaky(a_s[s * 4 + h] + a_d[d * 4 + h]);
        float mx = ord2f(nmax[d * 4 + h]);
        float ex = __expf(l - mx);
        alpha[(size_t)e * 4 + h] = ex;
        atomicAdd(&nsum[d * 4 + h], ex);
    }
}
__global__ void k_edge1_agg(const int* __restrict__ ei, int E, int Et,
                            const float* __restrict__ alpha, const float* __restrict__ nsum,
                            const float* __restrict__ h1, float* out1) {
    int id = blockIdx.x * blockDim.x + threadIdx.x;   // edge*4 + head
    if (id >= Et * 4) return;
    int e = id >> 2, h = id & 3;
    int s, d; edge_sd(ei, E, e, s, d);
    float coef = alpha[id] / (nsum[d * 4 + h] + 1e-16f);
    const float4* hs = reinterpret_cast<const float4*>(h1 + (size_t)s * D1 + h * 32);
    float* op = out1 + (size_t)d * D1 + h * 32;
#pragma unroll
    for (int q = 0; q < 8; ++q) {
        float4 v = hs[q];
        atomicAdd(op + q * 4 + 0, v.x * coef);
        atomicAdd(op + q * 4 + 1, v.y * coef);
        atomicAdd(op + q * 4 + 2, v.z * coef);
        atomicAdd(op + q * 4 + 3, v.w * coef);
    }
}

// ---------------- bias1 + BN + ReLU + GEMM2 + layer-2 projections -----------
__global__ __launch_bounds__(128) void k_post1(const float* __restrict__ out1,
        const float* __restrict__ bias1, const float* __restrict__ gamma,
        const float* __restrict__ beta,  const float* __restrict__ mean,
        const float* __restrict__ var,   const float* __restrict__ W2,
        const float* __restrict__ att_src2, const float* __restrict__ att_dst2,
        float* h2, float* a_src2, float* a_dst2, unsigned* nmax2, float* nsum2,
        float* d_out, int N) {
    __shared__ float sb[D1], sg[D1], sbt[D1], ssc[D1], sw[D1 * 3];
    int t = threadIdx.x;
    sb[t]  = bias1[t];
    sg[t]  = gamma[t];
    sbt[t] = beta[t];
    ssc[t] = rsqrtf(var[t] + 1e-5f);   // fold BN scale
    float mn = mean[t];
    sb[t] -= mn;                        // (v + bias1 - mean) in one add
    sw[t] = W2[t]; sw[D1 + t] = W2[D1 + t]; sw[2 * D1 + t] = W2[2 * D1 + t];
    __syncthreads();
    int n = blockIdx.x * 128 + t;
    if (n >= N) return;
    const float* row = out1 + (size_t)n * D1;
    float o0 = 0.f, o1 = 0.f, o2 = 0.f;
#pragma unroll 4
    for (int c = 0; c < D1; ++c) {
        float v = (row[c] + sb[c]) * ssc[c] * sg[c] + sbt[c];
        v = fmaxf(v, 0.f);
        o0 += v * sw[c * 3 + 0];
        o1 += v * sw[c * 3 + 1];
        o2 += v * sw[c * 3 + 2];
    }
    h2[(size_t)n * 3 + 0] = o0;
    h2[(size_t)n * 3 + 1] = o1;
    h2[(size_t)n * 3 + 2] = o2;
    a_src2[n] = o0 * att_src2[0] + o1 * att_src2[1] + o2 * att_src2[2];
    a_dst2[n] = o0 * att_dst2[0] + o1 * att_dst2[1] + o2 * att_dst2[2];
    nmax2[n] = 0u;
    nsum2[n] = 0.f;
    d_out[(size_t)n * 3 + 0] = 0.f;
    d_out[(size_t)n * 3 + 1] = 0.f;
    d_out[(size_t)n * 3 + 2] = 0.f;
}

// ---------------- layer-2 edge passes (H=1, C=3) ----------------------------
__global__ void k_edge2_max(const int* __restrict__ ei, int E, int Et,
                            const float* __restrict__ a_s, const float* __restrict__ a_d,
                            unsigned* nmax) {
    int e = blockIdx.x * blockDim.x + threadIdx.x;
    if (e >= Et) return;
    int s, d; edge_sd(ei, E, e, s, d);
    atomicMax(&nmax[d], f2ord(leaky(a_s[s] + a_d[d])));
}
__global__ void k_edge2_exp(const int* __restrict__ ei, int E, int Et,
                            const float* __restrict__ a_s, const float* __restrict__ a_d,
                            const unsigned* __restrict__ nmax, float* nsum, float* alpha) {
    int e = blockIdx.x * blockDim.x + threadIdx.x;
    if (e >= Et) return;
    int s, d; edge_sd(ei, E, e, s, d);
    float l  = leaky(a_s[s] + a_d[d]);
    float ex = __expf(l - ord2f(nmax[d]));
    alpha[e] = ex;
    atomicAdd(&nsum[d], ex);
}
__global__ void k_edge2_agg(const int* __restrict__ ei, int E, int Et,
                            const float* __restrict__ alpha, const float* __restrict__ nsum,
                            const float* __restrict__ h2, float* d_out) {
    int e = blockIdx.x * blockDim.x + threadIdx.x;
    if (e >= Et) return;
    int s, d; edge_sd(ei, E, e, s, d);
    float coef = alpha[e] / (nsum[d] + 1e-16f);
    atomicAdd(&d_out[(size_t)d * 3 + 0], h2[(size_t)s * 3 + 0] * coef);
    atomicAdd(&d_out[(size_t)d * 3 + 1], h2[(size_t)s * 3 + 1] * coef);
    atomicAdd(&d_out[(size_t)d * 3 + 2], h2[(size_t)s * 3 + 2] * coef);
}
__global__ void k_bias2(float* d_out, const float* __restrict__ bias2, int N) {
    int id = blockIdx.x * blockDim.x + threadIdx.x;
    if (id < N * 3) d_out[id] += bias2[id % 3];
}

// ---------------- launcher ---------------------------------------------------
extern "C" void kernel_launch(void* const* d_in, const int* in_sizes, int n_in,
                              void* d_out, int out_size, void* d_ws, size_t ws_size,
                              hipStream_t stream) {
    const float* x        = (const float*)d_in[0];
    const int*   ei       = (const int*)d_in[1];       // [2, E] int32
    const float* W1       = (const float*)d_in[2];
    const float* att_src1 = (const float*)d_in[3];
    const float* att_dst1 = (const float*)d_in[4];
    const float* bias1    = (const float*)d_in[5];
    const float* gamma    = (const float*)d_in[6];
    const float* beta     = (const float*)d_in[7];
    const float* mean     = (const float*)d_in[8];
    const float* var      = (const float*)d_in[9];
    const float* W2       = (const float*)d_in[10];
    const float* att_src2 = (const float*)d_in[11];
    const float* att_dst2 = (const float*)d_in[12];
    const float* bias2    = (const float*)d_in[13];
    float* out = (float*)d_out;

    const int N  = in_sizes[0] / IN_CH;   // 50000
    const int E  = in_sizes[1] / 2;       // 800000
    const int Et = E + N;                 // with self loops

    char* w = (char*)d_ws;
    auto carve = [&](size_t bytes) -> char* {
        char* p = w; w += (bytes + 255) & ~(size_t)255; return p;
    };
    float*    h1     = (float*)   carve((size_t)N * D1 * 4);
    __bf16*   Bp     = (__bf16*)  carve((size_t)KT_N * 8 * 32 * 16 * 2);
    float*    a_s1   = (float*)   carve((size_t)N * 16);
    float*    a_d1   = (float*)   carve((size_t)N * 16);
    unsigned* nmax1  = (unsigned*)carve((size_t)N * 16);
    float*    nsum1  = (float*)   carve((size_t)N * 16);
    float*    alpha1 = (float*)   carve((size_t)Et * 16);
    float*    out1   = (float*)   carve((size_t)N * D1 * 4);
    float*    h2     = (float*)   carve((size_t)N * 12);
    float*    a_s2   = (float*)   carve((size_t)N * 4);
    float*    a_d2   = (float*)   carve((size_t)N * 4);
    unsigned* nmax2  = (unsigned*)carve((size_t)N * 4);
    float*    nsum2  = (float*)   carve((size_t)N * 4);
    float*    alpha2 = (float*)   carve((size_t)Et * 4);

    // Layer 1
    k_pack_w1<<<KT_N * 8, 32, 0, stream>>>(W1, Bp);
    k_zero_f32<<<(N * D1 + 255) / 256, 256, 0, stream>>>(out1, N * D1);
    k_gemm1<<<(N + 127) / 128, 256, 0, stream>>>(x, Bp, h1, N);
    k_proj1<<<(N * 4 + 255) / 256, 256, 0, stream>>>(h1, att_src1, att_dst1,
                                                     a_s1, a_d1, nmax1, nsum1, N);
    k_edge1_max<<<(Et + 255) / 256, 256, 0, stream>>>(ei, E, Et, a_s1, a_d1, nmax1);
    k_edge1_exp<<<(Et + 255) / 256, 256, 0, stream>>>(ei, E, Et, a_s1, a_d1,
                                                      nmax1, nsum1, alpha1);
    k_edge1_agg<<<(Et * 4 + 255) / 256, 256, 0, stream>>>(ei, E, Et, alpha1, nsum1,
                                                          h1, out1);
    // BN + ReLU + GEMM2 + layer-2 projections (+ init layer-2 accumulators)
    k_post1<<<(N + 127) / 128, 128, 0, stream>>>(out1, bias1, gamma, beta, mean, var,
                                                 W2, att_src2, att_dst2,
                                                 h2, a_s2, a_d2, nmax2, nsum2, out, N);
    // Layer 2
    k_edge2_max<<<(Et + 255) / 256, 256, 0, stream>>>(ei, E, Et, a_s2, a_d2, nmax2);
    k_edge2_exp<<<(Et + 255) / 256, 256, 0, stream>>>(ei, E, Et, a_s2, a_d2,
                                                      nmax2, nsum2, alpha2);
    k_edge2_agg<<<(Et + 255) / 256, 256, 0, stream>>>(ei, E, Et, alpha2, nsum2, h2, out);
    k_bias2<<<(N * 3 + 255) / 256, 256, 0, stream>>>(out, bias2, N);
}
